// DualStreamDTIClassifier_10144712753227
// MI455X (gfx1250) — matrix operand order repaired and checked by reading, not compiled, not run on hardware
//
#include <hip/hip_runtime.h>
#include <hip/hip_bf16.h>
#include <math.h>

// MI455X / gfx1250, wave32. All GEMM work on v_wmma_f32_16x16x32_f16.
//
// Roofline: ~24 GFLOP of matrix work after separable-term factorization of
// pair@W1; resident data < 4 MB (L2 = 192 MB) -> compute bound on WMMA.
// Weight fragments staged once per block into LDS (192 KB of the 320 KB WGP
// LDS) via async global->LDS loads, shared by all 8 waves.

typedef __attribute__((ext_vector_type(16))) _Float16 v16h;
typedef __attribute__((ext_vector_type(8)))  _Float16 v8h;
typedef __attribute__((ext_vector_type(8)))  float    v8f;
typedef __attribute__((ext_vector_type(4)))  int      v4i;

#define DD   256
#define HID  128
#define BB   2
#define LL   96
#define LP   768
#define TOPK 100
#define NEGV (-20.0f)

// LDS layout for k_pair_mlp (halfs): [0,65536) W1cd frags, [65536,81920) W2
// frags, [81920,98304) eight 16x128 h1 tiles.  196608 bytes dynamic LDS.
#define W1CD_H   65536
#define W2_OFF   65536
#define H1_OFF   81920
#define SMEM_BYTES ((size_t)(H1_OFF + 8 * 16 * HID) * 2)

#if defined(__HIP_DEVICE_COMPILE__) && \
    __has_builtin(__builtin_amdgcn_global_load_async_to_lds_b128) && \
    __has_builtin(__builtin_amdgcn_s_wait_asynccnt)
#define HAVE_ASYNC_LDS 1
#else
#define HAVE_ASYNC_LDS 0
#endif

__device__ __forceinline__ v8f zero8() {
    v8f z;
#pragma unroll
    for (int i = 0; i < 8; ++i) z[i] = 0.0f;
    return z;
}

__device__ __forceinline__ v16h make_frag(v8h lo, v8h hi) {
    return __builtin_shufflevector(lo, hi, 0, 1, 2, 3, 4, 5, 6, 7,
                                           8, 9, 10, 11, 12, 13, 14, 15);
}

__device__ __forceinline__ v8f wmma_f16(v16h a, v16h b, v8f c) {
    // D = A(16x32 f16) x B(32x16 f16) + C(16x16 f32)
    return __builtin_amdgcn_wmma_f32_16x16x32_f16(false, a, false, b, (short)0, c, false, false);
}

__device__ __forceinline__ float gelu_exact(float x) {
    return 0.5f * x * (1.0f + erff(x * 0.70710678118654752f));
}

// 16-byte global -> LDS copy; async (GLOBAL_LOAD_ASYNC_TO_LDS_B128) if the
// toolchain exposes it, plain VGPR round-trip otherwise.
__device__ __forceinline__ void copy16_g2l(const _Float16* g, _Float16* l) {
#if HAVE_ASYNC_LDS
    __builtin_amdgcn_global_load_async_to_lds_b128(
        (__attribute__((address_space(1))) v4i*)(g),
        (__attribute__((address_space(3))) v4i*)(l), 0, 0);
#else
    *(uint4*)l = *(const uint4*)g;
#endif
}

__device__ __forceinline__ void copy_fence() {
#if HAVE_ASYNC_LDS
    __builtin_amdgcn_s_wait_asynccnt(0);
#endif
    __syncthreads();
}

// ---------------------------------------------------------------- prep ----

__global__ void k_convert_f16(const float* __restrict__ src, _Float16* __restrict__ dst, int n) {
    int i = blockIdx.x * blockDim.x + threadIdx.x;
    if (i < n) dst[i] = (_Float16)src[i];
}

// Swizzle row-major W[K][N] (f32) into per-fragment, per-lane-contiguous f16:
// dst[((kt*Nt+nt)*32 + lane)*16 + t] = W[32*kt + 16*(t>>3) + 8*(lane>>4) + (t&7)][16*nt + (lane&15)]
// matching the CDNA5 WMMA B-matrix (32x16 f16) VGPR layout: each lane fetches
// its 16 halfs with two contiguous 16B loads.
__global__ void k_swizzle_w(const float* __restrict__ W, _Float16* __restrict__ dst, int K, int N) {
    int idx = blockIdx.x * blockDim.x + threadIdx.x;
    int n = K * N;
    if (idx >= n) return;
    int Nt = N >> 4;
    int t    = idx & 15;
    int lane = (idx >> 4) & 31;
    int f    = idx >> 9;
    int kt = f / Nt, nt = f % Nt;
    int kh = lane >> 4, nl = lane & 15;
    int row = (kt << 5) + ((t & 8) << 1) + (kh << 3) + (t & 7);
    int col = (nt << 4) + nl;
    dst[idx] = (_Float16)W[(size_t)row * N + col];
}

// ------------------------------------------------------------ wmma gemm ---
// One wave computes one 16x16 C tile.  A: row-major f16 [M][K].  Bswz: above.
// A-fragment (16x32 f16): lanes 0-15 row=m0+lane, halfs 0..7 = K k..k+7,
// halfs 8..15 = K k+16..k+23; lanes 16-31 same rows, K k+8..15 / k+24..31.
__global__ __launch_bounds__(256) void k_wmma_gemm(
    const _Float16* __restrict__ A, const _Float16* __restrict__ Bswz,
    const float* __restrict__ bias, void* __restrict__ Cout,
    int M, int K, int N, int out_f16) {
    const int wave = (int)((blockIdx.x * blockDim.x + threadIdx.x) >> 5);
    const int lane = threadIdx.x & 31;
    const int Mt = M >> 4, Nt = N >> 4, Kt = K >> 5;
    const int mt = wave / Nt, nt = wave % Nt;
    if (mt >= Mt) return;
    const int kh = lane >> 4, nl = lane & 15;
    const _Float16* arow = A + (size_t)((mt << 4) + nl) * K;

    v8f acc = zero8();
    for (int kt = 0; kt < Kt; ++kt) {
        const int kb = kt << 5;
        v8h a0 = *(const v8h*)(arow + kb + kh * 8);
        v8h a1 = *(const v8h*)(arow + kb + 16 + kh * 8);
        const _Float16* bp = Bswz + (((size_t)(kt * Nt + nt) * 32 + lane) << 4);
        v16h af = make_frag(a0, a1);
        v16h bf = make_frag(*(const v8h*)bp, *(const v8h*)(bp + 8));
        acc = wmma_f16(af, bf, acc);
    }
    const int col = (nt << 4) + nl;
    const float bv = bias ? bias[col] : 0.0f;
#pragma unroll
    for (int r = 0; r < 8; ++r) {
        const int rowOut = (mt << 4) + r + (kh << 3);
        const float v = acc[r] + bv;
        if (out_f16) ((_Float16*)Cout)[(size_t)rowOut * N + col] = (_Float16)v;
        else         ((float*)Cout)[(size_t)rowOut * N + col] = v;
    }
}

// -------------------------------------------------------- fused pair MLP --
// One wave handles one (b, i, 16 consecutive j) tile: M = 16 pairs.
// Layer-1 pairwise part: K = 512 (m = l*p over d, then a = |l-p| over d).
__global__ __launch_bounds__(256) void k_pair_mlp(
    const _Float16* __restrict__ l16, const _Float16* __restrict__ p16,
    const float* __restrict__ lA, const float* __restrict__ pB,
    const _Float16* __restrict__ W1swz,   // full 1024x128 swizzled (Nt=8)
    const _Float16* __restrict__ W2swz,   // 128x128 swizzled
    const float* __restrict__ b2, const float* __restrict__ W3,
    const float* __restrict__ b3,
    const unsigned char* __restrict__ lpad, const unsigned char* __restrict__ ppad,
    float* __restrict__ lmap) {
    extern __shared__ __align__(16) _Float16 smem[];

    const int wv   = threadIdx.x >> 5;
    const int lane = threadIdx.x & 31;
    const int kh = lane >> 4, nl = lane & 15;

    // ---- stage W1c/W1d (global frags 128..255) + W2 frags into LDS --------
    {
        const _Float16* g1 = W1swz + (size_t)128 * 512;   // frag 128 onward
#pragma unroll 1
        for (int idx = threadIdx.x; idx < (W1CD_H / 8) + (HID * HID / 8); idx += 256) {
            if (idx < W1CD_H / 8) copy16_g2l(g1 + (size_t)idx * 8, smem + (size_t)idx * 8);
            else {
                int j = idx - W1CD_H / 8;
                copy16_g2l(W2swz + (size_t)j * 8, smem + W2_OFF + (size_t)j * 8);
            }
        }
        copy_fence();
    }

    const int tile = blockIdx.x * 8 + wv;          // 9216 tiles exactly
    const int JT = LP / 16;                        // 48
    const int b   = tile / (LL * JT);
    const int rem = tile % (LL * JT);
    const int i   = rem / JT;
    const int j0  = (rem % JT) << 4;

    const _Float16* lrow = l16 + (size_t)(b * LL + i) * DD;
    const _Float16* prow = p16 + (size_t)(b * LP + j0 + nl) * DD;
    _Float16* h1t = smem + H1_OFF + wv * (16 * HID);

    v8f acc[8];
#pragma unroll
    for (int n = 0; n < 8; ++n) acc[n] = zero8();

#pragma unroll 1
    for (int ks = 0; ks < 8; ++ks) {
        const int kb = ks << 5;
        v8h L0 = *(const v8h*)(lrow + kb + kh * 8);
        v8h L1 = *(const v8h*)(lrow + kb + 16 + kh * 8);
        v8h P0 = *(const v8h*)(prow + kb + kh * 8);
        v8h P1 = *(const v8h*)(prow + kb + 16 + kh * 8);
        // packed f16: v_pk_mul / v_pk_sub + abs
        v8h m0 = L0 * P0, m1 = L1 * P1;
        v8h a0 = __builtin_elementwise_abs(L0 - P0);
        v8h a1 = __builtin_elementwise_abs(L1 - P1);
        v16h mf = make_frag(m0, m1);
        v16h af = make_frag(a0, a1);
#pragma unroll
        for (int n = 0; n < 8; ++n) {           // (le*pe) @ W1c   (local frags 0..63)
            const _Float16* bp = smem + ((((ks * 8 + n) * 32) + lane) << 4);
            v16h bf = make_frag(*(const v8h*)bp, *(const v8h*)(bp + 8));
            acc[n] = wmma_f16(mf, bf, acc[n]);
        }
#pragma unroll
        for (int n = 0; n < 8; ++n) {           // |le-pe| @ W1d  (local frags 64..127)
            const _Float16* bp = smem + ((((64 + ks * 8 + n) * 32) + lane) << 4);
            v16h bf = make_frag(*(const v8h*)bp, *(const v8h*)(bp + 8));
            acc[n] = wmma_f16(af, bf, acc[n]);
        }
    }

    // epilogue 1: + lA(i) (includes b1) + pB(j), exact GELU, f16 tile -> LDS
    const float* lArow = lA + (size_t)(b * LL + i) * HID;
#pragma unroll
    for (int n = 0; n < 8; ++n) {
        const int col = (n << 4) + nl;
        const float la = lArow[col];
#pragma unroll
        for (int r = 0; r < 8; ++r) {
            const int jl = r + (kh << 3);
            float v = acc[n][r] + la + pB[(size_t)(b * LP + j0 + jl) * HID + col];
            h1t[jl * HID + col] = (_Float16)gelu_exact(v);
        }
    }
    __syncthreads();

    // layer 2: h1(16x128) @ W2(128x128)
    v8f acc2[8];
#pragma unroll
    for (int n = 0; n < 8; ++n) acc2[n] = zero8();
#pragma unroll 1
    for (int ks = 0; ks < 4; ++ks) {
        const int kb = ks << 5;
        v8h H0 = *(const v8h*)(h1t + nl * HID + kb + kh * 8);
        v8h H1 = *(const v8h*)(h1t + nl * HID + kb + 16 + kh * 8);
        v16h a2 = make_frag(H0, H1);
#pragma unroll
        for (int n = 0; n < 8; ++n) {
            const _Float16* bp = smem + W2_OFF + ((((ks * 8 + n) * 32) + lane) << 4);
            v16h bf = make_frag(*(const v8h*)bp, *(const v8h*)(bp + 8));
            acc2[n] = wmma_f16(a2, bf, acc2[n]);
        }
    }

    // epilogue 2: +b2, GELU, dot with W3 (128 -> 1), reduce across 16 lanes
    float part[8];
#pragma unroll
    for (int r = 0; r < 8; ++r) part[r] = 0.0f;
#pragma unroll
    for (int n = 0; n < 8; ++n) {
        const int col = (n << 4) + nl;
        const float b2v = b2[col], w3v = W3[col];
#pragma unroll
        for (int r = 0; r < 8; ++r)
            part[r] += gelu_exact(acc2[n][r] + b2v) * w3v;
    }
#pragma unroll
    for (int m = 1; m <= 8; m <<= 1) {
#pragma unroll
        for (int r = 0; r < 8; ++r) part[r] += __shfl_xor(part[r], m, 32);
    }

    if (nl == 0) {                     // lanes 0 (rows 0..7) and 16 (rows 8..15)
        const bool ip = lpad[b * LL + i] != 0;
        const float b3v = b3[0];
#pragma unroll
        for (int r = 0; r < 8; ++r) {
            const int j = j0 + r + (kh << 3);
            float v = part[r] + b3v;
            if (v != v) v = 0.0f;                       // nan -> 0
            else if (v == INFINITY)  v = 20.0f;         // +inf -> 20
            else if (v == -INFINITY) v = NEGV;          // -inf -> -20
            if (ip || ppad[b * LP + j]) v = NEGV;       // padding mask
            lmap[(size_t)b * (LL * LP) + i * LP + j] = v;
        }
    }
}

// ------------------------------------------------------- top-k + softmax --
// Destructive iterative argmax over the (recomputed-every-launch) logit map.
__global__ __launch_bounds__(1024) void k_topk_softmax(float* __restrict__ lm, float* __restrict__ out) {
    const int b = blockIdx.x;
    float* base = lm + (size_t)b * (LL * LP);
    const int n = LL * LP;
    __shared__ float sv[1024];
    __shared__ int   si[1024];
    __shared__ float topv[TOPK];
    const int tid = threadIdx.x;

    for (int t = 0; t < TOPK; ++t) {
        float best = -INFINITY; int bi = 0;
        for (int idx = tid; idx < n; idx += 1024) {
            float v = base[idx];
            if (v > best) { best = v; bi = idx; }
        }
        sv[tid] = best; si[tid] = bi;
        __syncthreads();
        for (int s = 512; s > 0; s >>= 1) {
            if (tid < s && sv[tid + s] > sv[tid]) { sv[tid] = sv[tid + s]; si[tid] = si[tid + s]; }
            __syncthreads();
        }
        if (tid == 0) { topv[t] = sv[0]; base[si[0]] = -INFINITY; }
        __syncthreads();
    }
    if (tid == 0) {
        const float m = topv[0];                    // extraction order is descending
        float s = 0.0f;
        for (int t = 0; t < TOPK; ++t) s += expf(topv[t] - m);
        float accv = 0.0f;
        for (int t = 0; t < TOPK; ++t) accv += (expf(topv[t] - m) / s) * topv[t];
        out[b] = accv;
    }
}

// ------------------------------------------------------------- launcher ---

extern "C" void kernel_launch(void* const* d_in, const int* in_sizes, int n_in,
                              void* d_out, int out_size, void* d_ws, size_t ws_size,
                              hipStream_t stream) {
    const float* l_tok = (const float*)d_in[0];
    const float* p_tok = (const float*)d_in[1];
    const unsigned char* l_pad = (const unsigned char*)d_in[2];   // jnp bool = 1 byte
    const unsigned char* p_pad = (const unsigned char*)d_in[3];
    const float* Wl = (const float*)d_in[4];
    const float* bl = (const float*)d_in[5];
    const float* Wp = (const float*)d_in[6];
    const float* bp = (const float*)d_in[7];
    const float* W1 = (const float*)d_in[8];
    const float* b1 = (const float*)d_in[9];
    const float* W2 = (const float*)d_in[10];
    const float* b2 = (const float*)d_in[11];
    const float* W3 = (const float*)d_in[12];
    const float* b3 = (const float*)d_in[13];
    float* out = (float*)d_out;

    char* ws = (char*)d_ws;
    size_t off = 0;
    auto alloc = [&](size_t bytes) { char* p = ws + off; off += (bytes + 255) & ~(size_t)255; return p; };
    _Float16* tokL16 = (_Float16*)alloc((size_t)BB * LL * DD * 2);
    _Float16* tokP16 = (_Float16*)alloc((size_t)BB * LP * DD * 2);
    _Float16* WlS    = (_Float16*)alloc((size_t)DD * DD * 2);
    _Float16* WpS    = (_Float16*)alloc((size_t)DD * DD * 2);
    _Float16* W1S    = (_Float16*)alloc((size_t)4 * DD * HID * 2);
    _Float16* W2S    = (_Float16*)alloc((size_t)HID * HID * 2);
    _Float16* l16    = (_Float16*)alloc((size_t)BB * LL * DD * 2);
    _Float16* p16    = (_Float16*)alloc((size_t)BB * LP * DD * 2);
    float*    lA     = (float*)alloc((size_t)BB * LL * HID * 4);
    float*    pB     = (float*)alloc((size_t)BB * LP * HID * 4);
    float*    lmap   = (float*)alloc((size_t)BB * LL * LP * 4);
    (void)ws_size; (void)n_in; (void)in_sizes; (void)out_size;

    auto cdiv = [](int a, int b) { return (a + b - 1) / b; };

    // f32 -> f16 tokens
    k_convert_f16<<<cdiv(BB * LL * DD, 256), 256, 0, stream>>>(l_tok, tokL16, BB * LL * DD);
    k_convert_f16<<<cdiv(BB * LP * DD, 256), 256, 0, stream>>>(p_tok, tokP16, BB * LP * DD);
    // swizzle all weight matrices to WMMA B-fragment layout
    k_swizzle_w<<<cdiv(DD * DD, 256), 256, 0, stream>>>(Wl, WlS, DD, DD);
    k_swizzle_w<<<cdiv(DD * DD, 256), 256, 0, stream>>>(Wp, WpS, DD, DD);
    k_swizzle_w<<<cdiv(4 * DD * HID, 256), 256, 0, stream>>>(W1, W1S, 4 * DD, HID);
    k_swizzle_w<<<cdiv(HID * HID, 256), 256, 0, stream>>>(W2, W2S, HID, HID);

    // l = l_tok @ Wl + bl (f16 out), p = p_tok @ Wp + bp (f16 out)
    {
        int waves = (BB * LL / 16) * (DD / 16);
        k_wmma_gemm<<<cdiv(waves * 32, 256), 256, 0, stream>>>(tokL16, WlS, bl, l16, BB * LL, DD, DD, 1);
        waves = (BB * LP / 16) * (DD / 16);
        k_wmma_gemm<<<cdiv(waves * 32, 256), 256, 0, stream>>>(tokP16, WpS, bp, p16, BB * LP, DD, DD, 1);
    }
    // separable layer-1 terms: lA = l @ W1a + b1 ; pB = p @ W1b
    {
        const _Float16* W1aS = W1S;                                // kt 0..7
        const _Float16* W1bS = W1S + (size_t)8 * (HID / 16) * 512; // kt 8..15
        int waves = (BB * LL / 16) * (HID / 16);
        k_wmma_gemm<<<cdiv(waves * 32, 256), 256, 0, stream>>>(l16, W1aS, b1, lA, BB * LL, DD, HID, 0);
        waves = (BB * LP / 16) * (HID / 16);
        k_wmma_gemm<<<cdiv(waves * 32, 256), 256, 0, stream>>>(p16, W1bS, nullptr, pB, BB * LP, DD, HID, 0);
    }
    // fused pairwise MLP -> logit map  (9216 tiles, 8 waves/block, 192KB LDS)
    k_pair_mlp<<<(BB * LL * (LP / 16)) / 8, 256, SMEM_BYTES, stream>>>(
        l16, p16, lA, pB, W1S, W2S, b2, W3, b3, l_pad, p_pad, lmap);
    // top-100 + softmax-weighted sum per batch
    k_topk_softmax<<<BB, 1024, 0, stream>>>(lmap, out);
}
